// TernaryLinear_4604204941740
// MI455X (gfx1250) — matrix operand order repaired and checked by reading, not compiled, not run on hardware
//
#include <hip/hip_runtime.h>

// Ternary linear: y[m,n] = (sum_k x[m,k]*w[n,k]) * scale[n] + bias[n]
// M = 8*1024 = 8192, K = 4096, N = 16384.
// FP16 WMMA GEMM, double-buffered LDS staging.
// K-order inside each 32-wide WMMA block is permuted (0,2,1,3 per 4-group)
// identically for A and B tiles -> enables packed-16-bit int8->f16 weight
// conversion (v_pk shifts + v_cvt_f16_i16) with zero repacking movs.

typedef _Float16 half16 __attribute__((ext_vector_type(16)));
typedef _Float16 half8  __attribute__((ext_vector_type(8)));
typedef _Float16 half4  __attribute__((ext_vector_type(4)));
typedef float    float8 __attribute__((ext_vector_type(8)));
typedef float    f4v    __attribute__((ext_vector_type(4)));
typedef int      i4v    __attribute__((ext_vector_type(4)));
typedef short    s2v    __attribute__((ext_vector_type(2)));

#define M_DIM 8192
#define N_DIM 16384
#define K_DIM 4096
#define BM 128
#define BN 128
#define BK 32
#define LDK 40   // LDS row stride in halves: 32 + 8 pad -> 80B -> conflict-free

union FragU { half16 v; half8 h[2]; };

// 4 ternary int8 (one dword) -> 4 f16 in permuted K-order (b0,b2,b1,b3).
// Lowers to v_pk_lshlrev_b16 + 2x v_pk_ashrrev_i16 + 4x v_cvt_f16_i16.
__device__ __forceinline__ half4 cvt_w4(unsigned int d) {
    s2v s;
    __builtin_memcpy(&s, &d, 4);
    s2v e = (s2v)(s << 8) >> 8;   // sext(b0), sext(b2)
    s2v o = s >> 8;               // sext(b1), sext(b3)
    half4 h;
    h.x = (_Float16)e.x;          // pos 0 <- K+0
    h.y = (_Float16)e.y;          // pos 1 <- K+2
    h.z = (_Float16)o.x;          // pos 2 <- K+1
    h.w = (_Float16)o.y;          // pos 3 <- K+3
    return h;
}

// float4 (K+0..3) -> half4 in the same permuted order (h0,h2,h1,h3).
__device__ __forceinline__ half4 cvt_x4(f4v v) {
    half4 h;
    h.x = (_Float16)v.x;
    h.y = (_Float16)v.z;
    h.z = (_Float16)v.y;
    h.w = (_Float16)v.w;
    return h;
}

__global__ __launch_bounds__(256)
void ternary_linear_wmma(const float* __restrict__ x,
                         const signed char* __restrict__ wq,
                         const float* __restrict__ scale,
                         const float* __restrict__ bias,
                         float* __restrict__ y) {
    // Double-buffered staging: 2 * (10240 + 10240) = 40960 B LDS.
    __shared__ _Float16 As[2][BM * LDK];
    __shared__ _Float16 Bs[2][BN * LDK];

    const int tid  = threadIdx.x;
    const int lane = tid & 31;
    const int wave = tid >> 5;
    const int wm   = wave & 3;    // 4 waves along M (each 32 rows)
    const int wn   = wave >> 2;   // 2 waves along N (each 64 cols)
    const int blockM = blockIdx.y * BM;
    const int blockN = blockIdx.x * BN;

    // A staging: per pass 32 rows, one float4 (4 cols) per thread, 4 passes.
    const int a_row = tid >> 3;         // 0..31
    const int a_col = (tid & 7) << 2;   // 0,4,...,28
    // B staging: 16 int8 per thread, one pass covers 128 rows x 32 cols.
    const int b_row = tid >> 1;         // 0..127
    const int b_col = (tid & 1) << 4;   // 0 or 16

    // Fragment lane decomposition (K contiguous per lane half, same order for A and B).
    const int lrow = lane & 15;
    const int lhi  = lane >> 4;         // 0/1
    const int a_kb = lhi << 4;          // A: halves0-15 = pos kb..kb+15 (contiguous)
    const int b_kb = lhi << 4;          // B: halves0-15 = pos kb..kb+15 (contiguous)

    const float* xp       = x  + (size_t)(blockM + a_row) * K_DIM + a_col;
    const signed char* wp = wq + (size_t)(blockN + b_row) * K_DIM + b_col;

    float8 acc[2][4];
    #pragma unroll
    for (int i = 0; i < 2; ++i)
        #pragma unroll
        for (int j = 0; j < 4; ++j)
            acc[i][j] = (float8)0.0f;

    // ---------- prologue: stage tile 0 into buffer 0 ----------
    {
        #pragma unroll
        for (int p = 0; p < 4; ++p) {
            const int row = a_row + p * 32;
            f4v v = *(const f4v*)(xp + (size_t)p * 32 * K_DIM);
            *(half4*)(&As[0][row * LDK + a_col]) = cvt_x4(v);
        }
        i4v raw = *(const i4v*)(wp);
        union { i4v i; unsigned int u[4]; } u; u.i = raw;
        half8 h0, h1;
        *(half4*)&h0      = cvt_w4(u.u[0]);
        *((half4*)&h0 + 1) = cvt_w4(u.u[1]);
        *(half4*)&h1      = cvt_w4(u.u[2]);
        *((half4*)&h1 + 1) = cvt_w4(u.u[3]);
        *(half8*)(&Bs[0][b_row * LDK + b_col])     = h0;
        *(half8*)(&Bs[0][b_row * LDK + b_col + 8]) = h1;
    }
    __syncthreads();

    _Float16* curA = &As[0][0];
    _Float16* nxtA = &As[1][0];
    _Float16* curB = &Bs[0][0];
    _Float16* nxtB = &Bs[1][0];

    const int nIter = K_DIM / BK;   // 128
    for (int it = 0; it < nIter; ++it) {
        const int kNext = (it + 1) * BK;
        const bool hasNext = (kNext < K_DIM);   // uniform

        // ---- issue next tile's global loads early (in flight over compute) ----
        f4v av[4];
        i4v braw;
        if (hasNext) {
            #pragma unroll
            for (int p = 0; p < 4; ++p)
                av[p] = *(const f4v*)(xp + (size_t)p * 32 * K_DIM + kNext);
            braw = *(const i4v*)(wp + kNext);
            if (kNext + BK < K_DIM)
                __builtin_prefetch(xp + kNext + BK, 0, 1);   // global_prefetch, 2 tiles ahead
        }

        // ---- load fragments from current buffer ----
        FragU a[2], b[4];
        #pragma unroll
        for (int mi = 0; mi < 2; ++mi) {
            const _Float16* p = curA + (wm * 32 + mi * 16 + lrow) * LDK;
            a[mi].h[0] = *(const half8*)(p + a_kb);
            a[mi].h[1] = *(const half8*)(p + a_kb + 8);
        }
        #pragma unroll
        for (int ni = 0; ni < 4; ++ni) {
            const _Float16* p = curB + (wn * 64 + ni * 16 + lrow) * LDK;
            b[ni].h[0] = *(const half8*)(p + b_kb);
            b[ni].h[1] = *(const half8*)(p + b_kb + 8);
        }

        // ---- 8 WMMAs: 32x64 wave tile, K += 32 ----
        #pragma unroll
        for (int mi = 0; mi < 2; ++mi)
            #pragma unroll
            for (int ni = 0; ni < 4; ++ni)
                acc[mi][ni] = __builtin_amdgcn_wmma_f32_16x16x32_f16(
                    false, a[mi].v, false, b[ni].v,
                    (short)0, acc[mi][ni], false, false);

        // ---- convert + stage next tile into the other buffer ----
        if (hasNext) {
            #pragma unroll
            for (int p = 0; p < 4; ++p) {
                const int row = a_row + p * 32;
                *(half4*)(&nxtA[row * LDK + a_col]) = cvt_x4(av[p]);
            }
            union { i4v i; unsigned int u[4]; } u; u.i = braw;
            half8 h0, h1;
            *(half4*)&h0      = cvt_w4(u.u[0]);
            *((half4*)&h0 + 1) = cvt_w4(u.u[1]);
            *(half4*)&h1      = cvt_w4(u.u[2]);
            *((half4*)&h1 + 1) = cvt_w4(u.u[3]);
            *(half8*)(&nxtB[b_row * LDK + b_col])     = h0;
            *(half8*)(&nxtB[b_row * LDK + b_col + 8]) = h1;
        }

        __syncthreads();

        _Float16* t;
        t = curA; curA = nxtA; nxtA = t;
        t = curB; curB = nxtB; nxtB = t;
    }

    // ---- epilogue: y = acc*scale[n] + bias[n] ----
    // C/D layout: vgpr r -> row (m&7), lane = n + 16*(m>=8).
    #pragma unroll
    for (int mi = 0; mi < 2; ++mi) {
        #pragma unroll
        for (int ni = 0; ni < 4; ++ni) {
            const int n  = blockN + wn * 64 + ni * 16 + lrow;
            const int mb = blockM + wm * 32 + mi * 16 + lhi * 8;
            const float s  = scale[n];
            const float bv = bias[n];
            float8 c = acc[mi][ni];
            #pragma unroll
            for (int r = 0; r < 8; ++r)
                y[(size_t)(mb + r) * N_DIM + n] = c[r] * s + bv;
        }
    }
}

extern "C" void kernel_launch(void* const* d_in, const int* in_sizes, int n_in,
                              void* d_out, int out_size, void* d_ws, size_t ws_size,
                              hipStream_t stream) {
    const float*       x     = (const float*)d_in[0];
    const signed char* wq    = (const signed char*)d_in[1];
    const float*       scale = (const float*)d_in[2];
    const float*       bias  = (const float*)d_in[3];
    float*             y     = (float*)d_out;

    dim3 grid(N_DIM / BN, M_DIM / BM);   // (128, 64)
    ternary_linear_wmma<<<grid, dim3(256), 0, stream>>>(x, wq, scale, bias, y);
}
